// CausalSelfAttention_85873576116762
// MI455X (gfx1250) — matrix operand order repaired and checked by reading, not compiled, not run on hardware
//
#include <hip/hip_runtime.h>
#include <hip/hip_bf16.h>
#include <math.h>

#define B_  2
#define S_  2048
#define E_  1024
#define H_  16
#define D_  64
#define E3_ 3072

typedef __attribute__((ext_vector_type(16))) __bf16 v16bf;
typedef __attribute__((ext_vector_type(8)))  __bf16 v8bf;
typedef __attribute__((ext_vector_type(8)))  float  v8f;

__device__ __forceinline__ unsigned short f2bf(float f) {
  union { float f; unsigned u; } v; v.f = f;
  unsigned r = v.u + 0x7FFFu + ((v.u >> 16) & 1u);   // round-to-nearest-even
  return (unsigned short)(r >> 16);
}

__device__ __forceinline__ v16bf cat8(v8bf lo, v8bf hi) {
  return __builtin_shufflevector(lo, hi, 0,1,2,3,4,5,6,7,8,9,10,11,12,13,14,15);
}
// A-fragment (16x32 bf16): lane l<16 holds row l, K = 0..15; lane l+16: K = 16..31
// caller passes p = &row[k0 + hf*16]; we read p[0..15]
__device__ __forceinline__ v16bf ld_a16(const unsigned short* p) {
  return *(const v16bf*)p;
}
// B-fragment (32x16 bf16): lane holds one N column, 16 *contiguous* K values ->
// single 32-byte vector load, lands in 8 consecutive VGPRs
__device__ __forceinline__ v16bf ld_b16(const unsigned short* p) {
  return *(const v16bf*)p;
}
__device__ __forceinline__ v8f wmma_bf16(v16bf a, v16bf b, v8f c) {
  return __builtin_amdgcn_wmma_f32_16x16x32_bf16(false, a, false, b, (short)0, c, false, false);
}

// cross-half (lane <-> lane^16) exchange, identity mapping, pure VALU
__device__ __forceinline__ unsigned xhalf_u(unsigned x) {
  return (unsigned)__builtin_amdgcn_permlanex16(
      (int)x, (int)x, 0x76543210, (int)0xFEDCBA98, false, false);
}
__device__ __forceinline__ float xhalf_f(float x) {
  union { float f; unsigned u; } a, b; a.f = x;
  b.u = xhalf_u(a.u);
  return b.f;
}
// pack two f32 -> packed bf16 pair (lo in [15:0], hi in [31:16])
__device__ __forceinline__ unsigned pk_bf2(float lo, float hi) {
  __hip_bfloat162 t = __float22bfloat162_rn(make_float2(lo, hi));
  union { __hip_bfloat162 b; unsigned u; } c; c.b = t;
  return c.u;
}

// ---------------- fp32 -> bf16 elementwise ----------------
__global__ void convert_x_kernel(const float* __restrict__ in,
                                 unsigned short* __restrict__ out, int n) {
  int i = blockIdx.x * blockDim.x + threadIdx.x;
  if (i < n) out[i] = f2bf(in[i]);
}

// ---------- transpose + convert: w[K][N] fp32 -> wt[N][K] bf16 ----------
__global__ void transpose_conv_kernel(const float* __restrict__ w,
                                      unsigned short* __restrict__ wt,
                                      int K, int N) {
  __shared__ float tile[32][33];
  int k0 = blockIdx.y * 32, n0 = blockIdx.x * 32;
  int tx = threadIdx.x, ty = threadIdx.y;                   // 32 x 8
  #pragma unroll
  for (int i = ty; i < 32; i += 8)
    tile[i][tx] = w[(size_t)(k0 + i) * N + n0 + tx];
  __syncthreads();
  #pragma unroll
  for (int i = ty; i < 32; i += 8)
    wt[(size_t)(n0 + i) * K + k0 + tx] = f2bf(tile[tx][i]);
}

// ---------- V slice of qkv -> vt[b][h][d][s] (bf16) ----------
__global__ void vtrans_kernel(const unsigned short* __restrict__ qkv,
                              unsigned short* __restrict__ vt) {
  __shared__ unsigned short tile[32][33];
  int bh = blockIdx.x; int b = bh >> 4, h = bh & 15;
  int s0 = blockIdx.y * 32, d0 = blockIdx.z * 32;
  int tx = threadIdx.x, ty = threadIdx.y;
  #pragma unroll
  for (int i = ty; i < 32; i += 8)
    tile[i][tx] = qkv[(size_t)(b * S_ + s0 + i) * E3_ + 2 * E_ + h * D_ + d0 + tx];
  __syncthreads();
  #pragma unroll
  for (int i = ty; i < 32; i += 8)
    vt[((size_t)bh * D_ + d0 + i) * S_ + s0 + tx] = tile[tx][i];
}

__device__ __forceinline__ void st_out(unsigned short* p, float v) { *p = f2bf(v); }
__device__ __forceinline__ void st_out(float* p, float v) { *p = v; }

// ---------- GEMM: C[M][N] = A[M][K](bf16) * Bt[N][K](bf16); 64x64 tile / wave ----------
template <typename OUT_T>
__global__ __launch_bounds__(128)
void gemm_wmma_kernel(const unsigned short* __restrict__ A,
                      const unsigned short* __restrict__ Bt,
                      OUT_T* __restrict__ C, int M, int N, int K) {
  int wave = (int)((blockIdx.x * blockDim.x + threadIdx.x) >> 5);
  int lane = threadIdx.x & 31;
  int ntiles = N / 64;
  int mt = wave / ntiles, nt = wave % ntiles;
  int mbase = mt * 64, nbase = nt * 64;
  if (mbase >= M) return;
  int l = lane & 15, hf = lane >> 4;

  v8f acc[4][4];
  #pragma unroll
  for (int i = 0; i < 4; ++i)
    #pragma unroll
    for (int j = 0; j < 4; ++j) { v8f z = {}; acc[i][j] = z; }

  const unsigned short* ap = A + (size_t)(mbase + l) * K + hf * 16;
  const unsigned short* bp = Bt + (size_t)(nbase + l) * K + hf * 16;

  for (int k0 = 0; k0 < K; k0 += 32) {
    v16bf a[4], bb[4];
    #pragma unroll
    for (int i = 0; i < 4; ++i)
      a[i] = ld_a16(ap + (size_t)i * 16 * K + k0);
    #pragma unroll
    for (int j = 0; j < 4; ++j)
      bb[j] = ld_b16(bp + (size_t)j * 16 * K + k0);
    #pragma unroll
    for (int i = 0; i < 4; ++i)
      #pragma unroll
      for (int j = 0; j < 4; ++j)
        acc[i][j] = wmma_bf16(a[i], bb[j], acc[i][j]);
  }
  #pragma unroll
  for (int i = 0; i < 4; ++i)
    #pragma unroll
    for (int r = 0; r < 8; ++r) {
      size_t base = (size_t)(mbase + i * 16 + hf * 8 + r) * N + nbase + l;
      #pragma unroll
      for (int j = 0; j < 4; ++j)
        st_out(&C[base + j * 16], acc[i][j][r]);
    }
}

// ---------- one kv-tile step of transposed flash attention ----------
// Computes S^T = K.Q^T (q in lanes, kv per-lane), softmax within lanes,
// builds P^T B-frag in registers, accumulates O^T = V^T.P^T.
template <bool MASKED>
__device__ __forceinline__ void attn_step(
    int kv0, int qbase, int l, int hf,
    const unsigned short* __restrict__ kbase,
    const unsigned short* __restrict__ vbase,
    v16bf qb[2][2], v8f ot[2][4], float mq[2], float lq[2]) {
  const float scale = 0.125f;                 // 1/sqrt(64)

  // K A-fragments: 2 kv-tiles x 2 d-halves
  v16bf ka[2][2];
  #pragma unroll
  for (int kvt = 0; kvt < 2; ++kvt) {
    const unsigned short* kp =
        kbase + (size_t)(kv0 + kvt * 16 + l) * E3_ + hf * 16;
    ka[kvt][0] = ld_a16(kp);
    ka[kvt][1] = ld_a16(kp + 32);
  }
  // V^T A-fragments: 4 d-tiles (K-dim = kv)
  v16bf va[4];
  #pragma unroll
  for (int dt = 0; dt < 4; ++dt)
    va[dt] = ld_a16(vbase + (size_t)(dt * 16 + l) * S_ + kv0 + hf * 16);

  #pragma unroll
  for (int qt = 0; qt < 2; ++qt) {
    // S^T tiles: rows = kv (per-lane), cols = q (lane l)
    v8f s0 = {}, s1 = {};
    s0 = wmma_bf16(ka[0][0], qb[qt][0], s0);
    s0 = wmma_bf16(ka[0][1], qb[qt][1], s0);
    s1 = wmma_bf16(ka[1][0], qb[qt][0], s1);
    s1 = wmma_bf16(ka[1][1], qb[qt][1], s1);

    float v0[8], v1[8];
    #pragma unroll
    for (int r = 0; r < 8; ++r) {
      v0[r] = s0[r] * scale;
      v1[r] = s1[r] * scale;
      if (MASKED) {
        int qg = qbase + qt * 16 + l;
        if (kv0 +      hf * 8 + r > qg) v0[r] = -__builtin_inff();
        if (kv0 + 16 + hf * 8 + r > qg) v1[r] = -__builtin_inff();
      }
    }
    // in-lane max over 16 kv values + one cross-half swap
    float mx = fmaxf(v0[0], v1[0]);
    #pragma unroll
    for (int r = 1; r < 8; ++r) mx = fmaxf(mx, fmaxf(v0[r], v1[r]));
    mx = fmaxf(mx, xhalf_f(mx));
    float mnew = fmaxf(mq[qt], mx);
    float alpha = __expf(mq[qt] - mnew);
    mq[qt] = mnew;

    float e0[8], e1[8];
    float rs = 0.f;
    #pragma unroll
    for (int r = 0; r < 8; ++r) {
      e0[r] = __expf(v0[r] - mnew);
      e1[r] = __expf(v1[r] - mnew);
      rs += e0[r] + e1[r];
    }
    rs += xhalf_f(rs);
    lq[qt] = lq[qt] * alpha + rs;

    // rescale O^T (per-lane scalar alpha)
    #pragma unroll
    for (int dt = 0; dt < 4; ++dt)
      #pragma unroll
      for (int r = 0; r < 8; ++r) ot[qt][dt][r] *= alpha;

    // Build P^T B-fragment (32 kv x 16 q) entirely in registers:
    // lanes hf=0 need K=0..15 (own kv 0..7 + partner 8..15),
    // lanes hf=1 need K=16..31 (partner 16..23 + own 24..31).
    unsigned pk0[4], pk1[4], sw0[4], sw1[4];
    #pragma unroll
    for (int i = 0; i < 4; ++i) {
      pk0[i] = pk_bf2(e0[2 * i], e0[2 * i + 1]);
      pk1[i] = pk_bf2(e1[2 * i], e1[2 * i + 1]);
      sw0[i] = xhalf_u(pk0[i]);
      sw1[i] = xhalf_u(pk1[i]);
    }
    union { unsigned u[8]; v16bf v; } pf;
    #pragma unroll
    for (int i = 0; i < 4; ++i) {
      pf.u[i]     = hf ? sw1[i] : pk0[i];
      pf.u[4 + i] = hf ? pk1[i] : sw0[i];
    }
    // O^T += V^T * P^T
    #pragma unroll
    for (int dt = 0; dt < 4; ++dt)
      ot[qt][dt] = wmma_bf16(va[dt], pf.v, ot[qt][dt]);
  }
}

// ---------- flash attention: one wave = 32 q rows of one (b,h); no LDS ----------
__global__ __launch_bounds__(128)
void attn_kernel(const unsigned short* __restrict__ qkv,
                 const unsigned short* __restrict__ vt,
                 unsigned short* __restrict__ y) {
  int wid  = threadIdx.x >> 5;
  int lane = threadIdx.x & 31;
  int gw   = blockIdx.x * 4 + wid;            // B*H*(S/32) = 2048 waves total
  int qt_ = gw & 63, bh = gw >> 6, h = bh & 15, b = bh >> 4;
  int qbase = qt_ * 32;
  int l = lane & 15, hf = lane >> 4;

  // Q^T B-fragments: 2 q-tiles x 2 d-halves (lane holds one q column)
  v16bf qb[2][2];
  #pragma unroll
  for (int qt = 0; qt < 2; ++qt) {
    const unsigned short* qp =
        qkv + (size_t)(b * S_ + qbase + qt * 16 + l) * E3_ + h * D_ + hf * 16;
    qb[qt][0] = ld_b16(qp);
    qb[qt][1] = ld_b16(qp + 32);
  }

  float mq[2], lq[2];
  v8f ot[2][4];
  #pragma unroll
  for (int qt = 0; qt < 2; ++qt) {
    mq[qt] = -__builtin_inff(); lq[qt] = 0.f;
    #pragma unroll
    for (int dt = 0; dt < 4; ++dt) { v8f z = {}; ot[qt][dt] = z; }
  }

  const unsigned short* kbase = qkv + (size_t)(b * S_) * E3_ + E_ + h * D_;
  const unsigned short* vbase = vt + (size_t)bh * D_ * S_;

  // bulk: fully-unmasked kv tiles [0, qbase)
  for (int kv0 = 0; kv0 < qbase; kv0 += 32)
    attn_step<false>(kv0, qbase, l, hf, kbase, vbase, qb, ot, mq, lq);
  // boundary: the single diagonal tile [qbase, qbase+32) needs causal masking
  attn_step<true>(qbase, qbase, l, hf, kbase, vbase, qb, ot, mq, lq);

  // epilogue: O^T fragments hold 8 contiguous d per lane -> packed b128 stores
  #pragma unroll
  for (int qt = 0; qt < 2; ++qt) {
    float inv = 1.0f / lq[qt];
    unsigned short* yp =
        y + (size_t)(b * S_ + qbase + qt * 16 + l) * E_ + h * D_ + hf * 8;
    #pragma unroll
    for (int dt = 0; dt < 4; ++dt) {
      uint4 w;
      w.x = pk_bf2(ot[qt][dt][0] * inv, ot[qt][dt][1] * inv);
      w.y = pk_bf2(ot[qt][dt][2] * inv, ot[qt][dt][3] * inv);
      w.z = pk_bf2(ot[qt][dt][4] * inv, ot[qt][dt][5] * inv);
      w.w = pk_bf2(ot[qt][dt][6] * inv, ot[qt][dt][7] * inv);
      *(uint4*)(yp + dt * 16) = w;
    }
  }
}

extern "C" void kernel_launch(void* const* d_in, const int* in_sizes, int n_in,
                              void* d_out, int out_size, void* d_ws, size_t ws_size,
                              hipStream_t stream) {
  const float* x      = (const float*)d_in[0];
  const float* w_attn = (const float*)d_in[1];
  const float* w_proj = (const float*)d_in[2];
  float* out = (float*)d_out;

  unsigned short* xb  = (unsigned short*)d_ws;            // 4M  : x bf16 [4096][1024]
  unsigned short* wat = xb  + (size_t)4  * 1024 * 1024;   // 3M  : w_attn^T bf16 [3072][1024]
  unsigned short* wpt = wat + (size_t)3  * 1024 * 1024;   // 1M  : w_proj^T bf16 [1024][1024]
  unsigned short* qkv = wpt + (size_t)1  * 1024 * 1024;   // 12M : qkv bf16 [4096][3072]
  unsigned short* vt  = qkv + (size_t)12 * 1024 * 1024;   // 4M  : v^T bf16 [32][64][2048]
  unsigned short* y   = vt  + (size_t)4  * 1024 * 1024;   // 4M  : attn out bf16 [4096][1024]

  const int nx = B_ * S_ * E_;
  convert_x_kernel<<<(nx + 255) / 256, 256, 0, stream>>>(x, xb, nx);

  dim3 tb(32, 8);
  transpose_conv_kernel<<<dim3(E3_ / 32, E_ / 32), tb, 0, stream>>>(w_attn, wat, E_, E3_);
  transpose_conv_kernel<<<dim3(E_ / 32, E_ / 32), tb, 0, stream>>>(w_proj, wpt, E_, E_);

  // QKV GEMM: M=4096, N=3072, K=1024 -> (64 * 48) waves of 64x64 tiles, 4 waves/block
  gemm_wmma_kernel<unsigned short><<<(64 * 48) / 4, 128, 0, stream>>>(
      xb, wat, qkv, B_ * S_, E3_, E_);

  vtrans_kernel<<<dim3(B_ * H_, S_ / 32, D_ / 32), tb, 0, stream>>>(qkv, vt);

  // attention: B*H*(S/32) = 2048 waves, 4/block
  attn_kernel<<<(B_ * H_ * (S_ / 32)) / 4, 128, 0, stream>>>(qkv, vt, y);

  // proj GEMM: M=4096, N=1024, K=1024 -> fp32 out, (64 * 16) waves
  gemm_wmma_kernel<float><<<(64 * 16) / 4, 128, 0, stream>>>(
      y, wpt, out, B_ * S_, E_, E_);
}
